// ANN_LSTM_67422396612617
// MI455X (gfx1250) — matrix operand order repaired and checked by reading, not compiled, not run on hardware
//
#include <hip/hip_runtime.h>

typedef _Float16 half_t;
typedef _Float16 v16h __attribute__((ext_vector_type(16)));
typedef _Float16 v8h  __attribute__((ext_vector_type(8)));
typedef float    v8f  __attribute__((ext_vector_type(8)));

#define HID     128
#define G4      512            // 4*HID gate columns
#define BCHUNK  16             // batch rows per workgroup (== WMMA M)
#define NWAVES  16
#define THREADS (NWAVES * 32)

// workspace layout (bytes): f16 weight copies + fused biases
#define WS_WH1   0                          // W_hh1 f16 [512][128]
#define WS_WI2   (G4 * HID * 2)             // W_ih2 f16
#define WS_WH2   (2 * G4 * HID * 2)         // W_hh2 f16
#define WS_BIAS1 (3 * G4 * HID * 2)         // b_ih1+b_hh1, 512 f32
#define WS_BIAS2 (WS_BIAS1 + G4 * 4)        // b_ih2+b_hh2, 512 f32

// Branch-free fast activations (exact limits at +/-inf, no libm branches):
//   sigmoid(x) = 1 / (1 + e^-x)
//   tanh(x)    = 1 - 2 / (e^{2x} + 1)
__device__ __forceinline__ float fsigm(float v) {
    return __builtin_amdgcn_rcpf(1.0f + __expf(-v));
}
__device__ __forceinline__ float ftanh(float v) {
    return 1.0f - 2.0f * __builtin_amdgcn_rcpf(__expf(2.0f * v) + 1.0f);
}

// One-time weight conversion fp32 -> f16 (row-major [512][128]) + bias fusion.
__global__ void lstm_prep(const float* __restrict__ Whh1, const float* __restrict__ Wih2,
                          const float* __restrict__ Whh2,
                          const float* __restrict__ bih1, const float* __restrict__ bhh1,
                          const float* __restrict__ bih2, const float* __restrict__ bhh2,
                          half_t* __restrict__ wh1, half_t* __restrict__ wi2,
                          half_t* __restrict__ wh2,
                          float* __restrict__ bias1, float* __restrict__ bias2) {
    int i = blockIdx.x * blockDim.x + threadIdx.x;
    if (i < G4 * HID) {
        wh1[i] = (half_t)Whh1[i];
        wi2[i] = (half_t)Wih2[i];
        wh2[i] = (half_t)Whh2[i];
    }
    if (i < G4) {
        bias1[i] = bih1[i] + bhh1[i];
        bias2[i] = bih2[i] + bhh2[i];
    }
}

// Build A fragment (16x32 f16, M=batch rows, K=hidden) from an LDS h-state tile.
// Lanes 0-15 (M=lane): K = [kb..kb+15]
// Lanes 16-31 (M=lane-16): K = [kb+16..kb+31]
__device__ __forceinline__ v16h load_a(const half_t* hs, int M, int kh, int kb) {
    return *(const v16h*)&hs[M * HID + kb + kh * 16];
}

// B fragment (32x16 f16, K x N) of W^T: B(k,n) = W[n][k]; per-lane 16 contiguous f16.
__device__ __forceinline__ v16h load_b(const half_t* __restrict__ w, int nrow, int kh, int kb) {
    return *(const v16h*)&w[nrow * HID + kb + kh * 16];
}

__global__ __launch_bounds__(THREADS, 1)
void lstm_persistent(const float* __restrict__ x, const float* __restrict__ Wih1,
                     const half_t* __restrict__ wh1, const half_t* __restrict__ wi2,
                     const half_t* __restrict__ wh2,
                     const float* __restrict__ bias1, const float* __restrict__ bias2,
                     const float* __restrict__ Wlin, const float* __restrict__ blin,
                     float* __restrict__ out, int T) {
    __shared__ float  gates[BCHUNK * G4];       // 32 KB: fp32 gate pre-activations
    __shared__ float  c1s[BCHUNK * HID];        // 8 KB
    __shared__ float  c2s[BCHUNK * HID];        // 8 KB
    __shared__ half_t h1s[BCHUNK * HID];        // 4 KB  (A operand, f16)
    __shared__ half_t h2s[BCHUNK * HID];        // 4 KB
    __shared__ float  wi1L[G4];                 // staged W_ih1 column (2 KB)
    __shared__ float  b1L[G4];                  // staged fused bias 1 (2 KB)
    __shared__ float  b2L[G4];                  // staged fused bias 2 (2 KB)
    __shared__ float  wlinL[HID];               // staged W_lin (0.5 KB)
    __shared__ float  xbuf[BCHUNK];
    __shared__ float  outacc[BCHUNK];

    const int tid   = threadIdx.x;
    const int lane  = tid & 31;
    const int wv    = tid >> 5;
    const int bbase = blockIdx.x * BCHUNK;      // global batch row base

    // zero-init recurrent state + stage loop-invariant vectors into LDS
    for (int i = tid; i < BCHUNK * HID; i += THREADS) {
        c1s[i] = 0.0f; c2s[i] = 0.0f;
        h1s[i] = (half_t)0.0f; h2s[i] = (half_t)0.0f;
    }
    if (tid < G4) { wi1L[tid] = Wih1[tid]; b1L[tid] = bias1[tid]; b2L[tid] = bias2[tid]; }
    if (tid < HID) wlinL[tid] = Wlin[tid];
    if (tid < BCHUNK) outacc[tid] = 0.0f;
    __syncthreads();

    const int M   = lane & 15;                  // A row / C column within N-tile
    const int kh  = lane >> 4;                  // K-half select
    const int nr0 = wv * 32 + (lane & 15);      // B row for N-tile 0 (tile 1 = +16)
    const int nc  = wv * 32 + (lane & 15);      // C/D column

    // elementwise indexing: 4 contiguous hidden elements per thread
    const int eb = (tid * 4) >> 7;              // batch row
    const int eh = (tid * 4) & 127;             // hidden base

    const float blin0 = blin[0];

    for (int t = 0; t < T; ++t) {
        if (tid < BCHUNK) xbuf[tid] = x[(bbase + tid) * T + t];

        // ============ GEMM 1: gates = h1 @ W_hh1^T ============
        v8f acc0 = {}, acc1 = {};
        #pragma unroll
        for (int kb = 0; kb < HID; kb += 32) {
            v16h a  = load_a(h1s, M, kh, kb);
            v16h b0 = load_b(wh1, nr0,      kh, kb);
            v16h b1 = load_b(wh1, nr0 + 16, kh, kb);
            acc0 = __builtin_amdgcn_wmma_f32_16x16x32_f16(false, a, false, b0,
                                                          (short)0, acc0, false, false);
            acc1 = __builtin_amdgcn_wmma_f32_16x16x32_f16(false, a, false, b1,
                                                          (short)0, acc1, false, false);
        }
        #pragma unroll
        for (int r = 0; r < 8; ++r) {
            int m = r + kh * 8;
            gates[m * G4 + nc]      = acc0[r];
            gates[m * G4 + nc + 16] = acc1[r];
        }
        __syncthreads();

        // ============ Elementwise LSTM cell 1 ============
        {
            float xv = xbuf[eb];
            #pragma unroll
            for (int j = 0; j < 4; ++j) {
                int hh = eh + j;
                float ig = gates[eb * G4 + hh]           + xv * wi1L[hh]           + b1L[hh];
                float fg = gates[eb * G4 + HID + hh]     + xv * wi1L[HID + hh]     + b1L[HID + hh];
                float gg = gates[eb * G4 + 2 * HID + hh] + xv * wi1L[2 * HID + hh] + b1L[2 * HID + hh];
                float og = gates[eb * G4 + 3 * HID + hh] + xv * wi1L[3 * HID + hh] + b1L[3 * HID + hh];
                float c = fsigm(fg) * c1s[eb * HID + hh] + fsigm(ig) * ftanh(gg);
                c1s[eb * HID + hh] = c;
                h1s[eb * HID + hh] = (half_t)(fsigm(og) * ftanh(c));
            }
        }
        __syncthreads();

        // ===== GEMM 2: gates = h1_new @ W_ih2^T + h2 @ W_hh2^T (fused accum) =====
        acc0 = (v8f){}; acc1 = (v8f){};
        #pragma unroll
        for (int kb = 0; kb < HID; kb += 32) {
            v16h a  = load_a(h1s, M, kh, kb);
            v16h b0 = load_b(wi2, nr0,      kh, kb);
            v16h b1 = load_b(wi2, nr0 + 16, kh, kb);
            acc0 = __builtin_amdgcn_wmma_f32_16x16x32_f16(false, a, false, b0,
                                                          (short)0, acc0, false, false);
            acc1 = __builtin_amdgcn_wmma_f32_16x16x32_f16(false, a, false, b1,
                                                          (short)0, acc1, false, false);
        }
        #pragma unroll
        for (int kb = 0; kb < HID; kb += 32) {
            v16h a  = load_a(h2s, M, kh, kb);
            v16h b0 = load_b(wh2, nr0,      kh, kb);
            v16h b1 = load_b(wh2, nr0 + 16, kh, kb);
            acc0 = __builtin_amdgcn_wmma_f32_16x16x32_f16(false, a, false, b0,
                                                          (short)0, acc0, false, false);
            acc1 = __builtin_amdgcn_wmma_f32_16x16x32_f16(false, a, false, b1,
                                                          (short)0, acc1, false, false);
        }
        #pragma unroll
        for (int r = 0; r < 8; ++r) {
            int m = r + kh * 8;
            gates[m * G4 + nc]      = acc0[r];
            gates[m * G4 + nc + 16] = acc1[r];
        }
        __syncthreads();

        // ===== Elementwise LSTM cell 2 + partial dot with W_lin =====
        {
            float psum = 0.0f;
            #pragma unroll
            for (int j = 0; j < 4; ++j) {
                int hh = eh + j;
                float ig = gates[eb * G4 + hh]           + b2L[hh];
                float fg = gates[eb * G4 + HID + hh]     + b2L[HID + hh];
                float gg = gates[eb * G4 + 2 * HID + hh] + b2L[2 * HID + hh];
                float og = gates[eb * G4 + 3 * HID + hh] + b2L[3 * HID + hh];
                float c = fsigm(fg) * c2s[eb * HID + hh] + fsigm(ig) * ftanh(gg);
                c2s[eb * HID + hh] = c;
                float h = fsigm(og) * ftanh(c);
                h2s[eb * HID + hh] = (half_t)h;
                psum += h * wlinL[hh];
            }
            atomicAdd(&outacc[eb], psum);
        }
        __syncthreads();

        // ===== Emit per-step output, reset accumulator =====
        if (tid < BCHUNK) {
            out[(bbase + tid) * T + t] = outacc[tid] + blin0;
            outacc[tid] = 0.0f;
        }
        // next step's first __syncthreads (after GEMM1 store) orders these writes
    }
}

extern "C" void kernel_launch(void* const* d_in, const int* in_sizes, int n_in,
                              void* d_out, int out_size, void* d_ws, size_t ws_size,
                              hipStream_t stream) {
    const float* x    = (const float*)d_in[0];
    const float* Wih1 = (const float*)d_in[1];
    const float* bih1 = (const float*)d_in[2];
    const float* Whh1 = (const float*)d_in[3];
    const float* bhh1 = (const float*)d_in[4];
    const float* Wih2 = (const float*)d_in[5];
    const float* bih2 = (const float*)d_in[6];
    const float* Whh2 = (const float*)d_in[7];
    const float* bhh2 = (const float*)d_in[8];
    const float* Wlin = (const float*)d_in[9];
    const float* blin = (const float*)d_in[10];

    const int B = 256;
    const int T = in_sizes[0] / B;

    char* ws = (char*)d_ws;
    half_t* wh1   = (half_t*)(ws + WS_WH1);
    half_t* wi2   = (half_t*)(ws + WS_WI2);
    half_t* wh2   = (half_t*)(ws + WS_WH2);
    float*  bias1 = (float*)(ws + WS_BIAS1);
    float*  bias2 = (float*)(ws + WS_BIAS2);

    lstm_prep<<<(G4 * HID + 511) / 512, 512, 0, stream>>>(
        Whh1, Wih2, Whh2, bih1, bhh1, bih2, bhh2, wh1, wi2, wh2, bias1, bias2);

    lstm_persistent<<<B / BCHUNK, THREADS, 0, stream>>>(
        x, Wih1, wh1, wi2, wh2, bias1, bias2, Wlin, blin, (float*)d_out, T);
}
